// FirstNet_88536455840401
// MI455X (gfx1250) — matrix operand order repaired and checked by reading, not compiled, not run on hardware
//
#include <hip/hip_runtime.h>
#include <hip/hip_bf16.h>

// ---------------- CDNA5 WMMA types ----------------
typedef __attribute__((ext_vector_type(16))) __bf16 v16bf;
typedef __attribute__((ext_vector_type(8)))  float  v8f;
typedef __attribute__((ext_vector_type(4)))  float  f4;
typedef __attribute__((ext_vector_type(4)))  int    v4i;

#if defined(__HIP_DEVICE_COMPILE__) && __has_builtin(__builtin_amdgcn_global_load_async_to_lds_b128)
#define HAVE_ASYNC_LDS 1
// builtin signature (from hipcc diagnostic): param0 = int4 in AS1 ("__device__"),
// param1 = LDS-side pointer (we use int4 in AS3), then imm offset + cpol.
typedef __attribute__((address_space(1))) v4i* gptr_b128;
typedef __attribute__((address_space(3))) v4i* lptr_b128;
#else
#define HAVE_ASYNC_LDS 0
#endif

__device__ __forceinline__ unsigned short f2bf(float f) {
    union { float f; unsigned u; } x; x.f = f;
    unsigned r = x.u + 0x7FFFu + ((x.u >> 16) & 1u);   // round-to-nearest-even
    return (unsigned short)(r >> 16);
}

__device__ __forceinline__ void wait_async_lds() {
#if HAVE_ASYNC_LDS
#if __has_builtin(__builtin_amdgcn_s_wait_asynccnt)
    __builtin_amdgcn_s_wait_asynccnt(0);
#else
    asm volatile("s_wait_asynccnt 0" ::: "memory");
#endif
#endif
}

static inline int cdiv(int a, int b) { return (a + b - 1) / b; }

// =====================================================================
// Generic grouped 3D conv as implicit GEMM with bf16 WMMA.
//   out[b, g*Cout+n, od,oh,ow] = sum_k A[m,k]*B[k,n]
//   m = (b, od,oh,ow) flattened, k = ci*kvol + (kd,kh,kw)
// 2D convs pass Din=1, kd=1. Block = 128 thr = 4 waves; wave w owns M-tile
// blockIdx.x*4+w; the block owns a 32-wide N tile (two 16-wide WMMA tiles).
// B tile (32n x 32k, f32) staged in LDS: interior tiles via
// global_load_async_to_lds_b128 (ASYNCcnt), edge tiles via guarded stores.
// Each wave issues 2 WMMAs per K-step sharing one A (im2col) fragment.
// =====================================================================
struct ConvP {
    const float* in; const float* w; float* out;
    int B, Gn, Cin, Din, Hin, Win;
    int Cout, Dout, Hout, Wout;
    int kd, kh, kw;
    int sd, sh, sw;
    int pd, ph, pw;
    int dd, dh, dw;
};

__global__ __launch_bounds__(128) void conv_gemm_wmma(ConvP p) {
    const int spat = p.Dout * p.Hout * p.Wout;
    const int M    = p.B * spat;
    const int kvol = p.kd * p.kh * p.kw;
    const int K    = p.Cin * kvol;
    const int N    = p.Cout;

    const int lane = threadIdx.x & 31;
    const int wv   = threadIdx.x >> 5;
    const int h    = lane >> 4;       // half-wave
    const int ml   = lane & 15;       // row within tile (A) / col (B, D)

    const int mTile = (blockIdx.x * 4 + wv) * 16;
    const int nBase = blockIdx.y * 32;
    const int g     = blockIdx.z;

    __shared__ float Bs[32 * 32];     // [n][k] f32 tile (4 KB)

    // decode this lane's M row (same for both half-waves)
    int m = mTile + ml;
    bool mok = (m < M);
    int b = 0, od = 0, oh = 0, ow = 0;
    if (mok) {
        b = m / spat; int s = m - b * spat;
        od = s / (p.Hout * p.Wout); int s2 = s - od * p.Hout * p.Wout;
        oh = s2 / p.Wout; ow = s2 - oh * p.Wout;
    }
    const int idB = od * p.sd - p.pd;
    const int ihB = oh * p.sh - p.ph;
    const int iwB = ow * p.sw - p.pw;
    const long HW = (long)p.Hin * p.Win;
    const float* inB = p.in + ((long)b * (p.Gn * p.Cin) + (long)g * p.Cin) * ((long)p.Din * HW);
    const float* wG  = p.w + (long)g * N * K + (long)nBase * K;

    // cooperative B-tile mapping: thread covers 8 consecutive k of one n row
    const int e  = threadIdx.x * 8;
    const int nl = e >> 5;            // 0..31
    const int kl = e & 31;            // 0,8,16,24

    v8f acc0 = {0.f, 0.f, 0.f, 0.f, 0.f, 0.f, 0.f, 0.f};
    v8f acc1 = {0.f, 0.f, 0.f, 0.f, 0.f, 0.f, 0.f, 0.f};

    for (int kk = 0; kk < K; kk += 32) {
        const bool full = (nBase + 32 <= N) && (kk + 32 <= K);   // uniform over block
        const float* src = wG + (long)nl * K + kk + kl;
        if (full) {
#if HAVE_ASYNC_LDS
            __builtin_amdgcn_global_load_async_to_lds_b128(
                (gptr_b128)src, (lptr_b128)&Bs[nl * 32 + kl], 0, 0);
            __builtin_amdgcn_global_load_async_to_lds_b128(
                (gptr_b128)(src + 4), (lptr_b128)&Bs[nl * 32 + kl + 4], 0, 0);
#else
            *(f4*)&Bs[nl * 32 + kl]     = *(const f4*)src;
            *(f4*)&Bs[nl * 32 + kl + 4] = *(const f4*)(src + 4);
#endif
            if (kk + 32 < K) __builtin_prefetch(src + 32, 0, 1);   // global_prefetch
        } else {
            bool nok = (nBase + nl) < N;
#pragma unroll
            for (int j = 0; j < 8; ++j) {
                float v = 0.f;
                if (nok && (kk + kl + j) < K) v = src[j];
                Bs[nl * 32 + kl + j] = v;
            }
        }
        wait_async_lds();
        __syncthreads();

        // ---- B fragments: lane(col=ml), elem i -> K = kk + h*16 + i (contig in LDS)
        v16bf bf0, bf1;
        {
            const float* bp0 = &Bs[ml * 32 + h * 16];
            const float* bp1 = bp0 + 16 * 32;
#pragma unroll
            for (int i = 0; i < 16; ++i) {
                ((unsigned short*)&bf0)[i] = f2bf(bp0[i]);
                ((unsigned short*)&bf1)[i] = f2bf(bp1[i]);
            }
        }

        // ---- A fragment: elem i -> K = kk + h*16 + i  (im2col gather)
        v16bf afrag;
#pragma unroll
        for (int i = 0; i < 16; ++i) {
            int k = kk + h * 16 + i;
            float v = 0.f;
            if (mok && k < K) {
                int ci = k / kvol;  int r  = k - ci * kvol;
                int kdi = r / (p.kh * p.kw); int r2 = r - kdi * (p.kh * p.kw);
                int khi = r2 / p.kw; int kwi = r2 - khi * p.kw;
                int id = idB + kdi * p.dd;
                int ih = ihB + khi * p.dh;
                int iw = iwB + kwi * p.dw;
                if ((unsigned)id < (unsigned)p.Din && (unsigned)ih < (unsigned)p.Hin &&
                    (unsigned)iw < (unsigned)p.Win)
                    v = inB[((long)ci * p.Din + id) * HW + (long)ih * p.Win + iw];
            }
            ((unsigned short*)&afrag)[i] = f2bf(v);
        }

        acc0 = __builtin_amdgcn_wmma_f32_16x16x32_bf16(false, afrag, false, bf0,
                                                       (short)0, acc0, false, false);
        acc1 = __builtin_amdgcn_wmma_f32_16x16x32_bf16(false, afrag, false, bf1,
                                                       (short)0, acc1, false, false);
        __syncthreads();
    }

    // ---- store D: lane col n=ml(+16), VGPR r -> row mTile + r + 8*h ----
    int n0 = nBase + ml;
    int n1 = n0 + 16;
#pragma unroll
    for (int r = 0; r < 8; ++r) {
        int mo = mTile + r + 8 * h;
        if (mo < M) {
            int bo = mo / spat; int s = mo - bo * spat;
            long base = ((long)bo * (p.Gn * p.Cout) + (long)g * p.Cout) * spat + s;
            if (n0 < N) p.out[base + (long)n0 * spat] = acc0[r];
            if (n1 < N) p.out[base + (long)n1 * spat] = acc1[r];
        }
    }
}

// =====================================================================
// Backbone stage 0: streaming BN stats over conv output (not materialized)
// conv: [4,1,96^3] -> [4,32,94^3], k=3, p=0, d=1, s=1
// =====================================================================
__global__ __launch_bounds__(512) void bb0_stats(const float* __restrict__ x,
                                                 const float* __restrict__ w,
                                                 const float* __restrict__ gamma,
                                                 const float* __restrict__ beta,
                                                 float* __restrict__ ss) {
    const int c = blockIdx.x;                  // 0..31
    const int O = 94, I = 96;
    const long total = 4L * O * O * O;
    float wl[27];
#pragma unroll
    for (int i = 0; i < 27; ++i) wl[i] = w[c * 27 + i];
    float sum = 0.f, sq = 0.f;
    for (long idx = threadIdx.x; idx < total; idx += blockDim.x) {
        long t = idx;
        int ow = (int)(t % O); t /= O;
        int oh = (int)(t % O); t /= O;
        int od = (int)(t % O); int b = (int)(t / O);
        const float* ip = x + (long)b * I * I * I + (long)od * I * I + (long)oh * I + ow;
        float a = 0.f;
#pragma unroll
        for (int kd = 0; kd < 3; ++kd)
#pragma unroll
            for (int kh = 0; kh < 3; ++kh)
#pragma unroll
                for (int kw = 0; kw < 3; ++kw)
                    a += ip[(long)kd * I * I + kh * I + kw] * wl[kd * 9 + kh * 3 + kw];
        sum += a; sq += a * a;
    }
    __shared__ float s1[512], s2[512];
    s1[threadIdx.x] = sum; s2[threadIdx.x] = sq;
    __syncthreads();
    for (int st = 256; st > 0; st >>= 1) {
        if ((int)threadIdx.x < st) { s1[threadIdx.x] += s1[threadIdx.x + st];
                                     s2[threadIdx.x] += s2[threadIdx.x + st]; }
        __syncthreads();
    }
    if (threadIdx.x == 0) {
        float n  = (float)total;
        float mu = s1[0] / n;
        float va = s2[0] / n - mu * mu;        // biased var (jnp.var default)
        float sc = gamma[c] * rsqrtf(va + 1e-5f);
        ss[c * 2] = sc; ss[c * 2 + 1] = beta[c] - mu * sc;
    }
}

// fused conv0 + BN(scale/shift) + ReLU + maxpool(3, s=2): out [4,32,46^3]
__global__ void bb0_pool(const float* __restrict__ x, const float* __restrict__ w,
                         const float* __restrict__ ss, float* __restrict__ out) {
    const int P = 46, I = 96;
    const long total = 4L * 32 * P * P * P;
    for (long idx = (long)blockIdx.x * blockDim.x + threadIdx.x; idx < total;
         idx += (long)gridDim.x * blockDim.x) {
        long t = idx;
        int px = (int)(t % P); t /= P;
        int py = (int)(t % P); t /= P;
        int pz = (int)(t % P); t /= P;
        int c  = (int)(t % 32); int b = (int)(t / 32);
        float sc = ss[c * 2], sh = ss[c * 2 + 1];
        const float* wc = w + c * 27;
        float mx = -3.4e38f;
        for (int dz = 0; dz < 3; ++dz)
            for (int dy = 0; dy < 3; ++dy)
                for (int dx = 0; dx < 3; ++dx) {
                    int oz = pz * 2 + dz, oy = py * 2 + dy, ox = px * 2 + dx;
                    const float* ip = x + (long)b * I * I * I + (long)oz * I * I + (long)oy * I + ox;
                    float a = 0.f;
#pragma unroll
                    for (int kd = 0; kd < 3; ++kd)
#pragma unroll
                        for (int kh = 0; kh < 3; ++kh)
#pragma unroll
                            for (int kw = 0; kw < 3; ++kw)
                                a += ip[(long)kd * I * I + kh * I + kw] * wc[kd * 9 + kh * 3 + kw];
                    float v = a * sc + sh;
                    mx = fmaxf(mx, v);
                }
        out[idx] = fmaxf(mx, 0.f);            // relu commutes with max
    }
}

// =====================================================================
// Per-channel BN stats from a materialized buffer x[B, C, S] -> scale/shift
// =====================================================================
__global__ __launch_bounds__(256) void bn_stats(const float* __restrict__ x,
                                                const float* __restrict__ gamma,
                                                const float* __restrict__ beta,
                                                float* __restrict__ ss,
                                                int C, int B, long S) {
    const int c = blockIdx.x;
    const long per = (long)B * S;
    float sum = 0.f, sq = 0.f;
    for (long i = threadIdx.x; i < per; i += blockDim.x) {
        long b = i / S, s = i - b * S;
        float v = x[(b * C + c) * S + s];
        sum += v; sq += v * v;
    }
    __shared__ float s1[256], s2[256];
    s1[threadIdx.x] = sum; s2[threadIdx.x] = sq;
    __syncthreads();
    for (int st = 128; st > 0; st >>= 1) {
        if ((int)threadIdx.x < st) { s1[threadIdx.x] += s1[threadIdx.x + st];
                                     s2[threadIdx.x] += s2[threadIdx.x + st]; }
        __syncthreads();
    }
    if (threadIdx.x == 0) {
        float n  = (float)per;
        float mu = s1[0] / n;
        float va = s2[0] / n - mu * mu;
        float sc = gamma[c] * rsqrtf(va + 1e-5f);
        ss[c * 2] = sc; ss[c * 2 + 1] = beta[c] - mu * sc;
    }
}

// fused BN(scale/shift) + ReLU + maxpool (VALID, stride st)
struct PoolP {
    const float* x; const float* ss; float* out;
    int B, C, D, H, W;      // input dims
    int PD, PH, PW;         // output dims
    int kd, kh, kw, st;
};
__global__ void bn_relu_pool(PoolP q) {
    const long total = (long)q.B * q.C * q.PD * q.PH * q.PW;
    const long HW = (long)q.H * q.W;
    for (long idx = (long)blockIdx.x * blockDim.x + threadIdx.x; idx < total;
         idx += (long)gridDim.x * blockDim.x) {
        long t = idx;
        int px = (int)(t % q.PW); t /= q.PW;
        int py = (int)(t % q.PH); t /= q.PH;
        int pz = (int)(t % q.PD); t /= q.PD;
        int c  = (int)(t % q.C);  int b = (int)(t / q.C);
        float sc = q.ss[c * 2], sh = q.ss[c * 2 + 1];
        const float* base = q.x + ((long)b * q.C + c) * q.D * HW;
        float mx = -3.4e38f;
        for (int dz = 0; dz < q.kd; ++dz)
            for (int dy = 0; dy < q.kh; ++dy)
                for (int dx = 0; dx < q.kw; ++dx) {
                    int iz = pz * q.st + dz, iy = py * q.st + dy, ix = px * q.st + dx;
                    float v = base[(long)iz * HW + (long)iy * q.W + ix] * sc + sh;
                    mx = fmaxf(mx, v);
                }
        q.out[idx] = fmaxf(mx, 0.f);
    }
}

// mean over spatial: x[B,C,S] -> out[b*ldo + off + c]
__global__ void avg_spatial(const float* __restrict__ x, float* __restrict__ out,
                            int B, int C, int S, int ldo, int off) {
    int idx = blockIdx.x * blockDim.x + threadIdx.x;
    if (idx >= B * C) return;
    int b = idx / C, c = idx - b * C;
    const float* p = x + ((long)b * C + c) * S;
    float s = 0.f;
    for (int i = 0; i < S; ++i) s += p[i];
    out[(long)b * ldo + off + c] = s / (float)S;
}

// per-ROI fc: feat[B,90,128] x fcw[90,8,128] + fcb[90,8] -> out[B,720]
__global__ void branch_fc(const float* __restrict__ feat, const float* __restrict__ fcw,
                          const float* __restrict__ fcb, float* __restrict__ out) {
    int idx = blockIdx.x * blockDim.x + threadIdx.x;
    if (idx >= 4 * 720) return;
    int b = idx / 720, r = idx - b * 720, g = r >> 3, o = r & 7;
    const float* f  = feat + ((long)b * 90 + g) * 128;
    const float* wp = fcw + ((long)g * 8 + o) * 128;
    float s = fcb[g * 8 + o];
    for (int i = 0; i < 128; ++i) s += f[i] * wp[i];
    out[idx] = s;
}

// dense linear: in[B,I] x W[O,I] + bias -> out1[b*ld1+off1+o] (and optional out2)
__global__ void linear_k(const float* __restrict__ in, const float* __restrict__ W,
                         const float* __restrict__ bias,
                         float* out1, int ld1, int off1,
                         float* out2, int ld2, int off2,
                         int B, int I, int O) {
    int idx = blockIdx.x * blockDim.x + threadIdx.x;
    if (idx >= B * O) return;
    int b = idx / O, o = idx - b * O;
    const float* ip = in + (long)b * I;
    const float* wp = W + (long)o * I;
    float s = bias[o];
    for (int i = 0; i < I; ++i) s += ip[i] * wp[i];
    out1[(long)b * ld1 + off1 + o] = s;
    if (out2) out2[(long)b * ld2 + off2 + o] = s;
}

// =====================================================================
extern "C" void kernel_launch(void* const* d_in, const int* in_sizes, int n_in,
                              void* d_out, int out_size, void* d_ws, size_t ws_size,
                              hipStream_t stream) {
    (void)in_sizes; (void)n_in; (void)out_size; (void)ws_size;
    auto P = [&](int i) { return (const float*)d_in[i]; };
    const float* x1 = P(0);
    const float* xb[3] = { P(1), P(2), P(3) };      // [4,90,1,64,64] == [4,90,64,64]
    float* out = (float*)d_out;                     // 8 + 3*512 = 1544 f32

    // ---- workspace layout (floats); total ~225 MB, all fully rewritten per call
    float* ws  = (float*)d_ws;
    float* R1  = ws;                 // 12,459,008  backbone pool ping
    float* R2  = R1 + 12459008;      // 37,933,056  backbone conv pong
    float* R3  = R2 + 37933056;      //  2,949,120  branch conv ping
    float* R4  = R3 + 2949120;       //  2,767,680  branch pool pong
    float* R5  = R4 + 2767680;       //     46,080  branch feat [4,90,128]
    float* R6  = R5 + 46080;         //      2,880  branch fc out [4,720]
    float* CC  = R6 + 2880;          //      3,584  concat [4,896]
    float* FB1 = CC + 3584;          //      1,024  fc1 out [4,256]
    float* FB2 = FB1 + 1024;         //        256  fc2 out [4,64]
    float* SS  = FB2 + 256;          //     23,040  per-layer scale/shift (max 2*11520)

    auto conv = [&](const float* in, const float* w, float* o,
                    int B, int Gn, int Cin, int Din, int Hin, int Win,
                    int Cout, int Dout, int Hout, int Wout,
                    int kd, int kh, int kw, int sd, int sh, int sw,
                    int pd, int ph, int pw, int dd, int dh, int dw) {
        ConvP p{in, w, o, B, Gn, Cin, Din, Hin, Win, Cout, Dout, Hout, Wout,
                kd, kh, kw, sd, sh, sw, pd, ph, pw, dd, dh, dw};
        int M = B * Dout * Hout * Wout;
        dim3 grid(cdiv(cdiv(M, 16), 4), cdiv(Cout, 32), Gn);
        hipLaunchKernelGGL(conv_gemm_wmma, grid, dim3(128), 0, stream, p);
    };
    auto stats = [&](const float* x, const float* g, const float* b, int C, long S) {
        hipLaunchKernelGGL(bn_stats, dim3(C), dim3(256), 0, stream, x, g, b, SS, C, 4, S);
    };
    auto pool = [&](const float* x, float* o, int C, int D, int H, int W,
                    int PD, int PH, int PW, int kd, int kh, int kw, int st) {
        PoolP q{x, SS, o, 4, C, D, H, W, PD, PH, PW, kd, kh, kw, st};
        long tot = 4L * C * PD * PH * PW;
        int grid = (int)((tot + 255) / 256); if (grid > 32768) grid = 32768;
        hipLaunchKernelGGL(bn_relu_pool, dim3(grid), dim3(256), 0, stream, q);
    };

    // ================= backbone =================
    // stage 0: streaming stats + fused conv/BN/ReLU/pool -> R1 [4,32,46^3]
    hipLaunchKernelGGL(bb0_stats, dim3(32), dim3(512), 0, stream, x1, P(52), P(53), P(54), SS);
    hipLaunchKernelGGL(bb0_pool, dim3(16384), dim3(256), 0, stream, x1, P(52), SS, R1);
    // stage 1: conv 32->128, k3 p0 d2 -> R2 [4,128,42^3]; pool3 s2 -> R1 [4,128,20^3]
    conv(R1, P(55), R2, 4, 1, 32, 46, 46, 46, 128, 42, 42, 42, 3, 3, 3, 1, 1, 1, 0, 0, 0, 2, 2, 2);
    stats(R2, P(56), P(57), 128, 42L * 42 * 42);
    pool(R2, R1, 128, 42, 42, 42, 20, 20, 20, 3, 3, 3, 2);
    // stage 2: conv 128->256, k3 p2 d2 -> R2 [4,256,20^3]; pool3 s2 -> R1 [4,256,9^3]
    conv(R1, P(58), R2, 4, 1, 128, 20, 20, 20, 256, 20, 20, 20, 3, 3, 3, 1, 1, 1, 2, 2, 2, 2, 2, 2);
    stats(R2, P(59), P(60), 256, 20L * 20 * 20);
    pool(R2, R1, 256, 20, 20, 20, 9, 9, 9, 3, 3, 3, 2);
    // stage 3: conv 256->512, k2 p1 d2 -> R2 [4,512,9^3]; pool5 s2 -> R1 [4,512,3^3]
    conv(R1, P(61), R2, 4, 1, 256, 9, 9, 9, 512, 9, 9, 9, 2, 2, 2, 1, 1, 1, 1, 1, 1, 2, 2, 2);
    stats(R2, P(62), P(63), 512, 9L * 9 * 9);
    pool(R2, R1, 512, 9, 9, 9, 3, 3, 3, 5, 5, 5, 2);
    // mean over 3^3 -> concat cols [0,512)
    hipLaunchKernelGGL(avg_spatial, dim3(cdiv(4 * 512, 256)), dim3(256), 0, stream,
                       R1, CC, 4, 512, 27, 896, 0);

    // ================= branches (G=90 grouped convs via implicit GEMM) =========
    static const int BC[4][2] = { {1, 8}, {8, 32}, {32, 64}, {64, 128} };
    static const int HIN[4] = { 64, 31, 15, 7 };
    static const int HOUT[4] = { 32, 16, 8, 4 };
    for (int j = 0; j < 3; ++j) {
        int bi = 4 + 16 * j;                         // w0,g0,b0,...,w3,g3,b3,fcw,fcb,attw,attb
        const float* cur = xb[j];
        for (int s = 0; s < 4; ++s) {
            int cin = BC[s][0], cout = BC[s][1];
            int hin = HIN[s], hout = HOUT[s];
            conv(cur, P(bi + 3 * s), R3, 4, 90, cin, 1, hin, hin, cout, 1, hout, hout,
                 1, 3, 3, 1, 2, 2, 0, 2, 2, 1, 2, 2);
            stats(R3, P(bi + 3 * s + 1), P(bi + 3 * s + 2), 90 * cout, (long)hout * hout);
            pool(R3, R4, 90 * cout, 1, hout, hout, 1, hout - 1, hout - 1, 1, 2, 2, 1);
            cur = R4;
        }
        // adaptive avgpool over 3x3 -> feat [4,90,128]
        hipLaunchKernelGGL(avg_spatial, dim3(cdiv(4 * 11520, 256)), dim3(256), 0, stream,
                           R4, R5, 4, 11520, 9, 11520, 0);
        // per-ROI fc1 (128->8) -> [4,720]
        hipLaunchKernelGGL(branch_fc, dim3(cdiv(4 * 720, 256)), dim3(256), 0, stream,
                           R5, P(bi + 12), P(bi + 13), R6);
        // attention fc (720->128) -> concat cols [512+128j) AND d_out att_j
        hipLaunchKernelGGL(linear_k, dim3(cdiv(4 * 128, 128)), dim3(128), 0, stream,
                           R6, P(bi + 14), P(bi + 15),
                           CC, 896, 512 + 128 * j,
                           out + 8 + 512 * j, 128, 0,
                           4, 720, 128);
    }

    // ================= head FCs =================
    hipLaunchKernelGGL(linear_k, dim3(cdiv(4 * 256, 128)), dim3(128), 0, stream,
                       CC, P(64), P(65), FB1, 256, 0, (float*)nullptr, 0, 0, 4, 896, 256);
    hipLaunchKernelGGL(linear_k, dim3(cdiv(4 * 64, 128)), dim3(128), 0, stream,
                       FB1, P(66), P(67), FB2, 64, 0, (float*)nullptr, 0, 0, 4, 256, 64);
    hipLaunchKernelGGL(linear_k, dim3(1), dim3(128), 0, stream,
                       FB2, P(68), P(69), out, 2, 0, (float*)nullptr, 0, 0, 4, 64, 2);
}
